// CrossAttentionNew_52982716563981
// MI455X (gfx1250) — compile-verified, matrix operand order checked by reading, not run on hardware
//
#include <hip/hip_runtime.h>

// ---------------------------------------------------------------------------
// CDNA5 / gfx1250 implementation of CrossAttentionNew scoring.
// Dominant work (stage-2 projections, ~290 GFLOP) runs on
// v_wmma_f32_16x16x32_bf16 with triple-buffered LDS tiles filled by the
// gfx1250 async copy engine (global_load_async_to_lds_b128 / ASYNCcnt,
// two tiles in flight, in-order completion -> s_wait_asynccnt <= 3).
// Attention-weighted sums (small K, prob-sensitive) use v_wmma_f32_16x16x4_f32.
// ---------------------------------------------------------------------------

typedef __attribute__((ext_vector_type(16))) __bf16 v16bf;
typedef __attribute__((ext_vector_type(8)))  float  v8f;
typedef __attribute__((ext_vector_type(2)))  float  v2f;

#define D_DIM 1024
#define B1 32
#define B2 32
#define R_DIM 36
#define T_DIM 32
#define NEGV (-1e9f)
#define SCALE (0.03125f)   // 1/sqrt(1024)

__device__ __forceinline__ unsigned short f2bf(float f) {
    unsigned int u = __float_as_uint(f);
    unsigned int r = u + 0x7FFFu + ((u >> 16) & 1u);   // round-nearest-even
    return (unsigned short)(r >> 16);
}
__device__ __forceinline__ float bf2f(unsigned short h) {
    return __uint_as_float(((unsigned int)h) << 16);
}

// LDS byte offset of a __shared__ object: flat LDS addresses carry the
// workgroup-relative offset in the low 32 bits on gfx1250.
__device__ __forceinline__ unsigned lds_off(const void* p) {
    return (unsigned)(unsigned long long)p;
}

// Async global->LDS 16-byte copy (GV mode), tracked by ASYNCcnt.
__device__ __forceinline__ void async_copy_b128(unsigned lds, const void* gaddr) {
    asm volatile("global_load_async_to_lds_b128 %0, %1, off"
                 :: "v"(lds), "v"((unsigned long long)gaddr)
                 : "memory");
}
// Wait until at most 3 async ops outstanding (newest tile may be pending).
__device__ __forceinline__ void wait_async3() {
#if __has_builtin(__builtin_amdgcn_s_wait_asynccnt)
    __builtin_amdgcn_s_wait_asynccnt(3);
#else
    asm volatile("s_wait_asynccnt 0x3" ::: "memory");
#endif
}
__device__ __forceinline__ void wait_async0() {
#if __has_builtin(__builtin_amdgcn_s_wait_asynccnt)
    __builtin_amdgcn_s_wait_asynccnt(0);
#else
    asm volatile("s_wait_asynccnt 0x0" ::: "memory");
#endif
}

union FragBF { v16bf v; uint4 q[2]; };

// ---------------------------------------------------------------------------
// float -> bf16 conversion, 8 elements / thread (b128 in/out)
// ---------------------------------------------------------------------------
__global__ void cvt_bf16_kernel(const float* __restrict__ in,
                                unsigned short* __restrict__ out, int n8) {
    int i = blockIdx.x * blockDim.x + threadIdx.x;
    if (i < n8) {
        const float4* p = reinterpret_cast<const float4*>(in) + (size_t)i * 2;
        float4 x = p[0], y = p[1];
        uint4 o;
        o.x = (unsigned)f2bf(x.x) | ((unsigned)f2bf(x.y) << 16);
        o.y = (unsigned)f2bf(x.z) | ((unsigned)f2bf(x.w) << 16);
        o.z = (unsigned)f2bf(y.x) | ((unsigned)f2bf(y.y) << 16);
        o.w = (unsigned)f2bf(y.z) | ((unsigned)f2bf(y.w) << 16);
        *reinterpret_cast<uint4*>(out + (size_t)i * 8) = o;
    }
}

// ---------------------------------------------------------------------------
// C[M,N] = scale * A[M,K] * B[N,K]^T   (A,B bf16 row-major, fp32 accumulate)
// Batched via A_bs/B_bs/C_bs element strides, grid.z = batch.
// Block: 256 threads (8 waves as 4x2), tile 128x64; wave tile 32x32 (2x2 WMMA).
// K stepped by 32; tiles stream into triple-buffered LDS via async copies,
// two tiles in flight; last two K-steps peeled so the hot loop is branch-free.
// Row loads clamped to Mreal/Nreal for padded tiles.
// ---------------------------------------------------------------------------
__global__ __launch_bounds__(256) void gemm_bf16_nt(
    const unsigned short* __restrict__ A, const unsigned short* __restrict__ B,
    float* __restrict__ Cf, unsigned short* __restrict__ Cb,
    int M, int N, int K, int lda, int ldb, int ldc,
    int Mreal, int Nreal,
    long long A_bs, long long B_bs, long long C_bs, float scale) {
    __shared__ alignas(16) unsigned short As[3][128][40];  // +8 halves pad
    __shared__ alignas(16) unsigned short Bs[3][64][40];

    const int tid  = threadIdx.x;
    const int lane = tid & 31;
    const int wid  = tid >> 5;
    const int wm   = wid >> 1;        // 0..3
    const int wn   = wid & 1;         // 0..1
    const int bM   = blockIdx.y * 128;
    const int bN   = blockIdx.x * 64;
    const int batch = blockIdx.z;

    const unsigned short* Ab = A + (size_t)batch * (size_t)A_bs;
    const unsigned short* Bb = B + (size_t)batch * (size_t)B_bs;

    // staging map: A tile = 512 16B chunks (2/thread), B tile = 256 (1/thread)
    const int ca  = tid * 2;
    const int ar0 = ca >> 2,        ac0 = (ca & 3) * 8;
    const int ar1 = (ca + 1) >> 2,  ac1 = ((ca + 1) & 3) * 8;
    const int br  = tid >> 2,       bc  = (tid & 3) * 8;

    int gar0 = bM + ar0; if (gar0 >= Mreal) gar0 = Mreal - 1;
    int gar1 = bM + ar1; if (gar1 >= Mreal) gar1 = Mreal - 1;
    int gbr  = bN + br;  if (gbr  >= Nreal) gbr  = Nreal - 1;

    const unsigned short* pa0 = Ab + (size_t)gar0 * lda + ac0;
    const unsigned short* pa1 = Ab + (size_t)gar1 * lda + ac1;
    const unsigned short* pb  = Bb + (size_t)gbr  * ldb + bc;

    const int m16 = lane & 15;
    const int kb  = lane >> 4;        // K-half select per ISA fragment layout

    v8f acc[2][2] = {};

    auto issue = [&](int k0, int buf) {
        async_copy_b128(lds_off(&As[buf][ar0][ac0]), pa0 + k0);
        async_copy_b128(lds_off(&As[buf][ar1][ac1]), pa1 + k0);
        async_copy_b128(lds_off(&Bs[buf][br][bc]),   pb  + k0);
    };

    auto compute = [&](int buf) {
        FragBF fa0, fa1, fb0, fb1;
        fa0.q[0] = *reinterpret_cast<const uint4*>(&As[buf][wm * 32 + m16][kb * 8]);
        fa0.q[1] = *reinterpret_cast<const uint4*>(&As[buf][wm * 32 + m16][16 + kb * 8]);
        fa1.q[0] = *reinterpret_cast<const uint4*>(&As[buf][wm * 32 + 16 + m16][kb * 8]);
        fa1.q[1] = *reinterpret_cast<const uint4*>(&As[buf][wm * 32 + 16 + m16][16 + kb * 8]);
        fb0.q[0] = *reinterpret_cast<const uint4*>(&Bs[buf][wn * 32 + m16][kb * 8]);
        fb0.q[1] = *reinterpret_cast<const uint4*>(&Bs[buf][wn * 32 + m16][16 + kb * 8]);
        fb1.q[0] = *reinterpret_cast<const uint4*>(&Bs[buf][wn * 32 + 16 + m16][kb * 8]);
        fb1.q[1] = *reinterpret_cast<const uint4*>(&Bs[buf][wn * 32 + 16 + m16][16 + kb * 8]);

        acc[0][0] = __builtin_amdgcn_wmma_f32_16x16x32_bf16(false, fa0.v, false, fb0.v,
                                                            (short)0, acc[0][0], false, false);
        acc[0][1] = __builtin_amdgcn_wmma_f32_16x16x32_bf16(false, fa0.v, false, fb1.v,
                                                            (short)0, acc[0][1], false, false);
        acc[1][0] = __builtin_amdgcn_wmma_f32_16x16x32_bf16(false, fa1.v, false, fb0.v,
                                                            (short)0, acc[1][0], false, false);
        acc[1][1] = __builtin_amdgcn_wmma_f32_16x16x32_bf16(false, fa1.v, false, fb1.v,
                                                            (short)0, acc[1][1], false, false);
    };

    const int nIter = K / 32;         // always >= 2 here (K = 1024)
    issue(0, 0);
    issue(32, 1);

    // steady state: branch-free, division-free, two tiles always in flight
    int bufc = 0;                     // buffer holding tile `it`
#pragma unroll 3
    for (int it = 0; it < nIter - 2; ++it) {
        wait_async3();                // oldest tile landed (in-order retire)
        __syncthreads();              // all copies visible; prev compute done
        int bufn = bufc + 2; if (bufn >= 3) bufn -= 3;
        issue((it + 2) * 32, bufn);
        compute(bufc);
        ++bufc; if (bufc == 3) bufc = 0;
    }
    // epilogue: tile nIter-2 (one tile still behind it in flight)
    wait_async3();
    __syncthreads();
    compute(bufc);
    ++bufc; if (bufc == 3) bufc = 0;
    // epilogue: tile nIter-1 (drain fully)
    wait_async0();
    __syncthreads();
    compute(bufc);

    float*          Cfb = Cf ? Cf + (size_t)batch * (size_t)C_bs : nullptr;
    unsigned short* Cbb = Cb ? Cb + (size_t)batch * (size_t)C_bs : nullptr;
#pragma unroll
    for (int sa = 0; sa < 2; ++sa) {
#pragma unroll
        for (int sb = 0; sb < 2; ++sb) {
            const int col = bN + wn * 32 + sb * 16 + m16;
#pragma unroll
            for (int i = 0; i < 8; ++i) {
                int row = bM + wm * 32 + sa * 16 + kb * 8 + i;  // C layout: VGPR i <-> row kb*8+i
                if (row < M && col < N) {
                    float v = acc[sa][sb][i] * scale;
                    if (Cfb) Cfb[(size_t)row * ldc + col] = v;
                    if (Cbb) Cbb[(size_t)row * ldc + col] = f2bf(v);
                }
            }
        }
    }
}

// ---------------------------------------------------------------------------
// softmax over 36-wide groups of s1 rows: s1[bt, a*36+r], softmax over r
// ---------------------------------------------------------------------------
__global__ void softmax_p1_kernel(float* __restrict__ s) {
    int gid = blockIdx.x * blockDim.x + threadIdx.x;   // 32768 groups
    int bt = gid >> 5, a = gid & 31;
    float* row = s + (size_t)bt * (B1 * R_DIM) + a * R_DIM;
    float m = -3.4e38f;
    for (int r = 0; r < R_DIM; ++r) m = fmaxf(m, row[r]);
    float sum = 0.f;
    for (int r = 0; r < R_DIM; ++r) { float e = __expf(row[r] - m); row[r] = e; sum += e; }
    float inv = 1.f / sum;
    for (int r = 0; r < R_DIM; ++r) row[r] *= inv;
}

// softmax over 32-wide groups of s2 rows with key mask: s2[ar, b*32+t]
__global__ void softmax_p2_kernel(float* __restrict__ s, const int* __restrict__ mask) {
    int gid = blockIdx.x * blockDim.x + threadIdx.x;   // 36864 groups
    int ar = gid >> 5, b = gid & 31;
    float* row = s + (size_t)ar * (B2 * T_DIM) + b * T_DIM;
    float m = -3.4e38f;
    for (int t = 0; t < T_DIM; ++t) {
        float v = (mask[b * T_DIM + t] == 0) ? NEGV : row[t];
        row[t] = v;
        m = fmaxf(m, v);
    }
    float sum = 0.f;
    for (int t = 0; t < T_DIM; ++t) { float e = __expf(row[t] - m); row[t] = e; sum += e; }
    float inv = 1.f / sum;
    for (int t = 0; t < T_DIM; ++t) row[t] *= inv;
}

// ---------------------------------------------------------------------------
// C = P[M,K] * V[K,1024]  (fp32 probs x fp32 values), K = 36 or 32.
// Uses v_wmma_f32_16x16x4_f32 (16x16 tile per wave, K stepped by 4).
// cmode 0: C row index = batch*C_bs + row*1024   (weighted_v1, batch = a)
// cmode 1: row=ar -> ((ar/36)*32+batch)*36+ar%36 (weighted_v2, batch = b)
// ---------------------------------------------------------------------------
__global__ __launch_bounds__(256) void attn_apply_kernel(
    const float* __restrict__ P, const float* __restrict__ V,
    unsigned short* __restrict__ Cb,
    int K, int lda, int A_boff, long long V_bs, long long C_bs, int cmode) {
    const int tid  = threadIdx.x;
    const int lane = tid & 31;
    const int wid  = tid >> 5;
    const int m16  = lane & 15;
    const int kb   = lane >> 4;
    const int batch = blockIdx.z;
    const int rowA = blockIdx.y * 16 + m16;
    const int col  = blockIdx.x * 128 + wid * 16 + m16;
    const float* Arow = P + (size_t)rowA * lda + batch * A_boff;
    const float* Vb   = V + (size_t)batch * (size_t)V_bs;

    float accs[8];
#if __has_builtin(__builtin_amdgcn_wmma_f32_16x16x4_f32)
    v8f acc = {};
    for (int k0 = 0; k0 < K; k0 += 4) {
        v2f a, b;
        a.x = Arow[k0 + 2 * kb];
        a.y = Arow[k0 + 2 * kb + 1];
        b.x = Vb[(size_t)(k0 + 2 * kb) * D_DIM + col];
        b.y = Vb[(size_t)(k0 + 2 * kb + 1) * D_DIM + col];
        acc = __builtin_amdgcn_wmma_f32_16x16x4_f32(false, a, false, b,
                                                    (short)0, acc, false, false);
    }
#pragma unroll
    for (int i = 0; i < 8; ++i) accs[i] = acc[i];
#else
#pragma unroll
    for (int i = 0; i < 8; ++i) accs[i] = 0.f;
    for (int k = 0; k < K; ++k) {
        float bv = Vb[(size_t)k * D_DIM + col];
#pragma unroll
        for (int i = 0; i < 8; ++i) {
            int r = blockIdx.y * 16 + kb * 8 + i;
            accs[i] += P[(size_t)r * lda + batch * A_boff + k] * bv;
        }
    }
#endif
#pragma unroll
    for (int i = 0; i < 8; ++i) {
        int r = blockIdx.y * 16 + kb * 8 + i;
        size_t off;
        if (cmode == 0) {
            off = (size_t)batch * (size_t)C_bs + (size_t)r * D_DIM + col;
        } else {
            int a_ = r / R_DIM, rr = r % R_DIM;
            off = ((size_t)(a_ * B2 + batch) * R_DIM + rr) * D_DIM + col;
        }
        Cb[off] = f2bf(accs[i]);
    }
}

// ---------------------------------------------------------------------------
// softmax over s (valid 36 of padded 48) + mean over r -> pbar1[ab,48]
// ---------------------------------------------------------------------------
__global__ void sf_reduce1_kernel(const float* __restrict__ sf, float* __restrict__ pbar) {
    int ab = blockIdx.x, t = threadIdx.x;
    __shared__ float sp[48];
    if (t < 48) sp[t] = 0.f;
    __syncthreads();
    if (t < R_DIM) {
        const float* row = sf + (size_t)ab * (48 * 48) + (size_t)t * 48;
        float m = -3.4e38f;
        for (int s = 0; s < R_DIM; ++s) m = fmaxf(m, row[s]);
        float sum = 0.f;
        for (int s = 0; s < R_DIM; ++s) sum += __expf(row[s] - m);
        float inv = 1.f / (sum * (float)R_DIM);
        for (int s = 0; s < R_DIM; ++s) atomicAdd(&sp[s], __expf(row[s] - m) * inv);
    }
    __syncthreads();
    if (t < 48) pbar[(size_t)ab * 48 + t] = sp[t];
}

// masked softmax over s (32) + mean over t -> pbar2[ab,32]
__global__ void sf_reduce2_kernel(const float* __restrict__ sf, float* __restrict__ pbar,
                                  const int* __restrict__ mask) {
    int ab = blockIdx.x, t = threadIdx.x;   // 32 threads
    int b = ab & 31;
    __shared__ float sp[32];
    sp[t] = 0.f;
    __syncthreads();
    const float* row = sf + (size_t)ab * (32 * 32) + (size_t)t * 32;
    float m = -3.4e38f;
    float vals[32];
    for (int s = 0; s < T_DIM; ++s) {
        float v = (mask[b * T_DIM + s] == 0) ? NEGV : row[s];
        vals[s] = v;
        m = fmaxf(m, v);
    }
    float sum = 0.f;
    for (int s = 0; s < T_DIM; ++s) sum += __expf(vals[s] - m);
    float inv = 1.f / (sum * (float)T_DIM);
    for (int s = 0; s < T_DIM; ++s) atomicAdd(&sp[s], __expf(vals[s] - m) * inv);
    __syncthreads();
    pbar[(size_t)ab * 32 + t] = sp[t];
}

// co[ab,d] = sum_{s<S} pbar[ab*pstride+s] * bf16(wv[(ab*S+s)*1024+d])
__global__ void co_mix_kernel(const float* __restrict__ pbar,
                              const unsigned short* __restrict__ wv,
                              float* __restrict__ co, int S, int pstride) {
    int gid = blockIdx.x * blockDim.x + threadIdx.x;  // 1M threads
    int ab = gid >> 10, d = gid & 1023;
    float acc = 0.f;
    for (int s = 0; s < S; ++s)
        acc += pbar[(size_t)ab * pstride + s] * bf2f(wv[((size_t)ab * S + s) * D_DIM + d]);
    co[(size_t)ab * D_DIM + d] = acc;
}

// score[ab] = <co1/(|co1|+eps), co2/(|co2|+eps)>
__global__ void score_kernel(const float* __restrict__ co1, const float* __restrict__ co2,
                             float* __restrict__ out) {
    int ab = blockIdx.x, t = threadIdx.x;
    __shared__ float sa[256], sb[256], sc[256];
    float a = 0.f, b = 0.f, c = 0.f;
    for (int d = t; d < D_DIM; d += 256) {
        float x = co1[(size_t)ab * D_DIM + d];
        float y = co2[(size_t)ab * D_DIM + d];
        a += x * x; b += y * y; c += x * y;
    }
    sa[t] = a; sb[t] = b; sc[t] = c;
    __syncthreads();
    for (int o = 128; o > 0; o >>= 1) {
        if (t < o) { sa[t] += sa[t + o]; sb[t] += sb[t + o]; sc[t] += sc[t + o]; }
        __syncthreads();
    }
    if (t == 0) out[ab] = sc[0] / ((sqrtf(sa[0]) + 1e-8f) * (sqrtf(sb[0]) + 1e-8f));
}

// ---------------------------------------------------------------------------
extern "C" void kernel_launch(void* const* d_in, const int* in_sizes, int n_in,
                              void* d_out, int out_size, void* d_ws, size_t ws_size,
                              hipStream_t stream) {
    (void)in_sizes; (void)n_in; (void)out_size; (void)ws_size;
    const float* v1 = (const float*)d_in[0];          // [32,36,1024]
    const float* v2 = (const float*)d_in[1];          // [32,32,1024]
    // d_in[2..9] = Wk1, Wk2, Wq1, Wq2, Wwq1, Wwq2, Wwk1, Wwk2
    const int* mask = (const int*)d_in[10];           // [32,32]

    const size_t M1 = (size_t)B1 * R_DIM;             // 1152
    const size_t M2 = (size_t)B2 * T_DIM;             // 1024
    const size_t AB = (size_t)B1 * B2;                // 1024

    char* w = (char*)d_ws;
    size_t used = 0;
    auto alloc = [&](size_t bytes) -> void* {
        void* p = w + used;
        used = (used + bytes + 255) & ~(size_t)255;
        return p;
    };

    unsigned short* v1b  = (unsigned short*)alloc(M1 * D_DIM * 2);
    unsigned short* v2b  = (unsigned short*)alloc(M2 * D_DIM * 2);
    unsigned short* Wb   = (unsigned short*)alloc(8 * (size_t)D_DIM * D_DIM * 2);
    unsigned short* k1b  = (unsigned short*)alloc(M1 * D_DIM * 2);
    unsigned short* q1b  = (unsigned short*)alloc(M1 * D_DIM * 2);
    unsigned short* k2b  = (unsigned short*)alloc(M2 * D_DIM * 2);
    unsigned short* q2b  = (unsigned short*)alloc(M2 * D_DIM * 2);
    float*          s1   = (float*)alloc(M2 * M1 * 4);                  // [bt, ar]
    float*          s2   = (float*)alloc(M1 * M2 * 4);                  // [ar, bt]
    unsigned short* wv1b = (unsigned short*)alloc(AB * T_DIM * D_DIM * 2);
    unsigned short* wv2b = (unsigned short*)alloc(AB * R_DIM * D_DIM * 2);
    unsigned short* wv2q = (unsigned short*)alloc(AB * R_DIM * D_DIM * 2);
    unsigned short* wv2k = (unsigned short*)alloc(AB * R_DIM * D_DIM * 2);
    unsigned short* wv1q = (unsigned short*)alloc(AB * T_DIM * D_DIM * 2);
    unsigned short* wv1k = (unsigned short*)alloc(AB * T_DIM * D_DIM * 2);
    float*          sf1  = (float*)alloc(AB * 48 * 48 * 4);
    float*          sf2  = (float*)alloc(AB * 32 * 32 * 4);
    float*          pb1  = (float*)alloc(AB * 48 * 4);
    float*          pb2  = (float*)alloc(AB * 32 * 4);
    float*          co1  = (float*)alloc(AB * D_DIM * 4);
    float*          co2  = (float*)alloc(AB * D_DIM * 4);

    auto cvt = [&](const float* in, unsigned short* out, size_t n) {
        size_t n8 = n / 8;
        cvt_bf16_kernel<<<dim3((unsigned)((n8 + 255) / 256)), dim3(256), 0, stream>>>(in, out, (int)n8);
    };
    cvt(v1, v1b, M1 * D_DIM);
    cvt(v2, v2b, M2 * D_DIM);
    for (int i = 0; i < 8; ++i)
        cvt((const float*)d_in[2 + i], Wb + (size_t)i * D_DIM * D_DIM, (size_t)D_DIM * D_DIM);

    auto gemm = [&](const unsigned short* A, const unsigned short* B, float* Cf,
                    unsigned short* Cb, int M, int N, int K, int lda, int ldb, int ldc,
                    int Mreal, int Nreal, long long Abs, long long Bbs, long long Cbs,
                    int batch, float scale) {
        dim3 g((unsigned)((N + 63) / 64), (unsigned)((M + 127) / 128), (unsigned)batch);
        gemm_bf16_nt<<<g, dim3(256), 0, stream>>>(A, B, Cf, Cb, M, N, K, lda, ldb, ldc,
                                                  Mreal, Nreal, Abs, Bbs, Cbs, scale);
    };

    // input projections: x @ W^T  (bf16 WMMA, write bf16)
    gemm(v1b, Wb + 0 * (size_t)D_DIM * D_DIM, nullptr, k1b, (int)M1, D_DIM, D_DIM,
         D_DIM, D_DIM, D_DIM, (int)M1, D_DIM, 0, 0, 0, 1, 1.0f);       // Wk1
    gemm(v2b, Wb + 1 * (size_t)D_DIM * D_DIM, nullptr, k2b, (int)M2, D_DIM, D_DIM,
         D_DIM, D_DIM, D_DIM, (int)M2, D_DIM, 0, 0, 0, 1, 1.0f);       // Wk2
    gemm(v1b, Wb + 2 * (size_t)D_DIM * D_DIM, nullptr, q1b, (int)M1, D_DIM, D_DIM,
         D_DIM, D_DIM, D_DIM, (int)M1, D_DIM, 0, 0, 0, 1, 1.0f);       // Wq1
    gemm(v2b, Wb + 3 * (size_t)D_DIM * D_DIM, nullptr, q2b, (int)M2, D_DIM, D_DIM,
         D_DIM, D_DIM, D_DIM, (int)M2, D_DIM, 0, 0, 0, 1, 1.0f);       // Wq2

    // s1[bt, ar] = q2·k1^T * scale ; s2[ar, bt] = q1·k2^T * scale
    gemm(q2b, k1b, s1, nullptr, (int)M2, (int)M1, D_DIM, D_DIM, D_DIM, (int)M1,
         (int)M2, (int)M1, 0, 0, 0, 1, SCALE);
    gemm(q1b, k2b, s2, nullptr, (int)M1, (int)M2, D_DIM, D_DIM, D_DIM, (int)M2,
         (int)M1, (int)M2, 0, 0, 0, 1, SCALE);

    softmax_p1_kernel<<<dim3(128), dim3(256), 0, stream>>>(s1);
    softmax_p2_kernel<<<dim3(144), dim3(256), 0, stream>>>(s2, mask);

    // weighted_v1[a][bt,d] = p1 @ v1[a]  (fp32 WMMA, K=36), batch = a
    attn_apply_kernel<<<dim3(8, (unsigned)(M2 / 16), B1), dim3(256), 0, stream>>>(
        s1, v1, wv1b, R_DIM, (int)M1, R_DIM,
        (long long)R_DIM * D_DIM, (long long)M2 * D_DIM, 0);
    // weighted_v2[ar per batch b] = p2 @ v2[b] (K=32), batch = b, scattered C rows
    attn_apply_kernel<<<dim3(8, (unsigned)(M1 / 16), B2), dim3(256), 0, stream>>>(
        s2, v2, wv2b, T_DIM, (int)M2, T_DIM,
        (long long)T_DIM * D_DIM, 0, 1);

    // stage-2 projections (the big GEMMs)
    gemm(wv2b, Wb + 5 * (size_t)D_DIM * D_DIM, nullptr, wv2q, (int)(AB * R_DIM), D_DIM,
         D_DIM, D_DIM, D_DIM, D_DIM, (int)(AB * R_DIM), D_DIM, 0, 0, 0, 1, 1.0f); // Wwq2
    gemm(wv2b, Wb + 7 * (size_t)D_DIM * D_DIM, nullptr, wv2k, (int)(AB * R_DIM), D_DIM,
         D_DIM, D_DIM, D_DIM, D_DIM, (int)(AB * R_DIM), D_DIM, 0, 0, 0, 1, 1.0f); // Wwk2
    gemm(wv1b, Wb + 4 * (size_t)D_DIM * D_DIM, nullptr, wv1q, (int)(AB * T_DIM), D_DIM,
         D_DIM, D_DIM, D_DIM, D_DIM, (int)(AB * T_DIM), D_DIM, 0, 0, 0, 1, 1.0f); // Wwq1
    gemm(wv1b, Wb + 6 * (size_t)D_DIM * D_DIM, nullptr, wv1k, (int)(AB * T_DIM), D_DIM,
         D_DIM, D_DIM, D_DIM, D_DIM, (int)(AB * T_DIM), D_DIM, 0, 0, 0, 1, 1.0f); // Wwk1

    // batched score GEMMs: sf1 (padded 48x48 over 36x36), sf2 (32x32)
    gemm(wv2q, wv2k, sf1, nullptr, 48, 48, D_DIM, D_DIM, D_DIM, 48, R_DIM, R_DIM,
         (long long)R_DIM * D_DIM, (long long)R_DIM * D_DIM, 48 * 48, (int)AB, SCALE);
    gemm(wv1q, wv1k, sf2, nullptr, 32, 32, D_DIM, D_DIM, D_DIM, 32, T_DIM, T_DIM,
         (long long)T_DIM * D_DIM, (long long)T_DIM * D_DIM, 32 * 32, (int)AB, SCALE);

    sf_reduce1_kernel<<<dim3((unsigned)AB), dim3(64), 0, stream>>>(sf1, pb1);
    sf_reduce2_kernel<<<dim3((unsigned)AB), dim3(32), 0, stream>>>(sf2, pb2, mask);

    // co_v1 = mean-softmax(sf1) @ weighted_v2 ; co_v2 likewise with weighted_v1
    co_mix_kernel<<<dim3(4096), dim3(256), 0, stream>>>(pb1, wv2b, co1, R_DIM, 48);
    co_mix_kernel<<<dim3(4096), dim3(256), 0, stream>>>(pb2, wv1b, co2, T_DIM, 32);

    score_kernel<<<dim3((unsigned)AB), dim3(256), 0, stream>>>(co1, co2, (float*)d_out);
}
